// DeltaNet_31877247271458
// MI455X (gfx1250) — matrix operand order repaired and checked by reading, not compiled, not run on hardware
//
#include <hip/hip_runtime.h>
#include <hip/hip_bf16.h>
#include <math.h>

typedef __bf16 bf16;
typedef __attribute__((ext_vector_type(16))) __bf16 v16bf;
typedef __attribute__((ext_vector_type(8)))  float  v8f;
typedef __attribute__((ext_vector_type(4))) unsigned int v4u;
typedef __attribute__((ext_vector_type(8))) int v8i;
typedef __attribute__((ext_vector_type(4))) int v4i;

#define BBATCH 2
#define LLEN   2048
#define DMODEL 1024
#define NHEAD  4
#define DKH    256
#define DVH    256
#define CONVK  4
#define CCH    32              // chunk length
#define NCH    (LLEN/CCH)      // 64 chunks
#define DVT    16              // dv slice per scan block
#define NDV    (DVH/DVT)       // 16
#define BL     (BBATCH*LLEN)   // 4096
#define BHL    (BBATCH*NHEAD*LLEN) // 16384

__device__ __forceinline__ float bf2f(bf16 x) { return (float)x; }
__device__ __forceinline__ bf16  f2bf(float x) { return (bf16)x; }

__device__ __forceinline__ v8f wmma_acc(v16bf a, v16bf b, v8f c) {
  // D = A(16x32 bf16) * B(32x16 bf16) + C(16x16 f32)
  return __builtin_amdgcn_wmma_f32_16x16x32_bf16(false, a, false, b,
                                                 (short)0, c, false, false);
}

// ---- CDNA5 wave32 WMMA operand layout helpers (ISA 7.12.2) ----
// A operand: 16x32 bf16, row-major source with leading dim lda.
__device__ __forceinline__ v16bf ld_a(const bf16* a, int lda, int lane) {
  int m = lane & 15, hk = (lane >> 4) & 1;
  v16bf r;
#pragma unroll
  for (int v = 0; v < 8; ++v) {
    int k0 = (hk << 4) + (v << 1);
    r[2 * v]     = a[m * lda + k0];
    r[2 * v + 1] = a[m * lda + k0 + 1];
  }
  return r;
}
// A operand from f32 source (convert on load)
__device__ __forceinline__ v16bf ld_a_f32(const float* a, int lda, int lane) {
  int m = lane & 15, hk = (lane >> 4) & 1;
  v16bf r;
#pragma unroll
  for (int v = 0; v < 8; ++v) {
    int k0 = (hk << 4) + (v << 1);
    r[2 * v]     = f2bf(a[m * lda + k0]);
    r[2 * v + 1] = f2bf(a[m * lda + k0 + 1]);
  }
  return r;
}
// A operand where logical A[m][k] = src[k*ld + m]  (transposed source, for k^T)
__device__ __forceinline__ v16bf ld_at(const bf16* at, int ldat, int lane) {
  int m = lane & 15, hk = (lane >> 4) & 1;
  v16bf r;
#pragma unroll
  for (int v = 0; v < 8; ++v) {
    int k0 = (hk << 4) + (v << 1);
    r[2 * v]     = at[k0 * ldat + m];
    r[2 * v + 1] = at[(k0 + 1) * ldat + m];
  }
  return r;
}
// B operand: 32x16 bf16, row-major K x N source with leading dim ldb.
__device__ __forceinline__ v16bf ld_b(const bf16* b, int ldb, int lane) {
  int n = lane & 15, kb = ((lane >> 4) & 1) << 4;
  v16bf r;
#pragma unroll
  for (int e = 0; e < 16; ++e) r[e] = b[(kb + e) * ldb + n];
  return r;
}
__device__ __forceinline__ v16bf ld_b_f32(const float* b, int ldb, int lane) {
  int n = lane & 15, kb = ((lane >> 4) & 1) << 4;
  v16bf r;
#pragma unroll
  for (int e = 0; e < 16; ++e) r[e] = f2bf(b[(kb + e) * ldb + n]);
  return r;
}
// B operand where logical B[k][n] = src[n*ld + k]  (transposed source, for k^T)
__device__ __forceinline__ v16bf ld_bt(const bf16* bt, int ldt, int lane) {
  int n = lane & 15, kb = ((lane >> 4) & 1) << 4;
  v16bf r;
#pragma unroll
  for (int e = 0; e < 16; ++e) r[e] = bt[n * ldt + kb + e];
  return r;
}
// C operand: 16x16 f32 from row-major source.
__device__ __forceinline__ v8f ld_c(const float* c, int ldc, int lane) {
  int n = lane & 15, hk = (lane >> 4) & 1;
  v8f r;
#pragma unroll
  for (int v = 0; v < 8; ++v) r[v] = c[(v + hk * 8) * ldc + n];
  return r;
}

// ---- Tensor Data Mover: DMA a rows x cols f32 tile (row stride in elems)
// from global into LDS.  D# per CDNA5 ISA ch.8; 6-arg clang-23 builtin form.
__device__ __forceinline__ void tdm_load_tile_f32(const float* gsrc,
                                                  void* lds_dst,
                                                  int rows, int cols,
                                                  int row_stride) {
  unsigned long long ga = (unsigned long long)(uintptr_t)gsrc;
  unsigned int lds = (unsigned int)(uintptr_t)lds_dst;  // LDS byte offset
  v4u g0;
  g0[0] = 1u;                                        // count=1, user D#
  g0[1] = lds;                                       // lds_addr
  g0[2] = (unsigned int)ga;                          // global_addr[31:0]
  g0[3] = ((unsigned int)(ga >> 32) & 0x01FFFFFFu)   // global_addr[56:32]
          | (2u << 30);                              // type = 2 ("image")
  v8i g1;
  g1[0] = (int)(2u << 16);                           // wg_mask=0, data_size=4B
  g1[1] = (int)(((unsigned)row_stride & 0xFFFFu) << 16);      // tensor_dim0 lo
  g1[2] = (int)(((unsigned)row_stride >> 16) |
                (((unsigned)rows & 0xFFFFu) << 16)); // tensor_dim0 hi, dim1 lo
  g1[3] = (int)(((unsigned)cols & 0xFFFFu) << 16);   // dim1 hi=0, tile_dim0
  g1[4] = rows;                                      // tile_dim1 (tile_dim2=0)
  g1[5] = row_stride;                                // tensor_dim0_stride lo
  g1[6] = 0;                                         // stride hi, dim1_stride
  g1[7] = 0;
  v4i gz4 = {0, 0, 0, 0};
  v8i gz8 = {0, 0, 0, 0, 0, 0, 0, 0};
  __builtin_amdgcn_tensor_load_to_lds(g0, g1, gz4, gz4, gz8, 0);
}

// ================= GEMM: C[M,N](f32) = A[M,K] * B[K,N](f32 weights) =========
// A tile staged by threads (f32->bf16); B tile DMA'd by the TDM into a
// double-buffered f32 LDS tile, synchronized with TENSORcnt.
template <bool ABF16>
__global__ __launch_bounds__(256) void k_gemm(const void* __restrict__ Ap,
                                              const float* __restrict__ Bw,
                                              float* __restrict__ C,
                                              int M, int N, int K) {
  __shared__ bf16  As[128 * 32];
  __shared__ float Bs[2][32 * 64];
  int tid = threadIdx.x, lane = tid & 31, w = tid >> 5;
  int m0 = blockIdx.x * 128, n0 = blockIdx.y * 64;
  int wm = w >> 1, wn = w & 1;          // 4x2 wave grid, 32x32 per wave
  v8f a00 = {}, a01 = {}, a10 = {}, a11 = {};
  int nk = K / 32;
  if (w == 0)  // prologue DMA: first B tile
    tdm_load_tile_f32(Bw + n0, &Bs[0][0], 32, 64, N);
  for (int it = 0; it < nk; ++it) {
    int k0 = it * 32;
    int cur = it & 1;
    for (int i = tid; i < 128 * 32; i += 256) {
      int r = i >> 5, c = i & 31;
      if constexpr (ABF16)
        As[i] = ((const bf16*)Ap)[(size_t)(m0 + r) * K + k0 + c];
      else
        As[i] = f2bf(((const float*)Ap)[(size_t)(m0 + r) * K + k0 + c]);
    }
    if (w == 0) {
      __builtin_amdgcn_s_wait_tensorcnt(0);    // current B tile landed in LDS
      if (it + 1 < nk)                          // kick DMA for next B tile
        tdm_load_tile_f32(Bw + (size_t)(k0 + 32) * N + n0,
                          &Bs[cur ^ 1][0], 32, 64, N);
    }
    __syncthreads();
    v16bf fa0 = ld_a(As + (wm * 32) * 32, 32, lane);
    v16bf fa1 = ld_a(As + (wm * 32 + 16) * 32, 32, lane);
    v16bf fb0 = ld_b_f32(&Bs[cur][0] + wn * 32, 64, lane);
    v16bf fb1 = ld_b_f32(&Bs[cur][0] + wn * 32 + 16, 64, lane);
    a00 = wmma_acc(fa0, fb0, a00);
    a01 = wmma_acc(fa0, fb1, a01);
    a10 = wmma_acc(fa1, fb0, a10);
    a11 = wmma_acc(fa1, fb1, a11);
    __syncthreads();
  }
  int n = lane & 15, hk = (lane >> 4) & 1;
#pragma unroll
  for (int r = 0; r < 8; ++r) {
    int mm0 = m0 + wm * 32 + r + hk * 8;
    int mm1 = mm0 + 16;
    int nn0 = n0 + wn * 32 + n;
    C[(size_t)mm0 * N + nn0]      = a00[r];
    C[(size_t)mm0 * N + nn0 + 16] = a01[r];
    C[(size_t)mm1 * N + nn0]      = a10[r];
    C[(size_t)mm1 * N + nn0 + 16] = a11[r];
  }
}

// ================= gates: beta_l, beta_g, w_l, w_g, w_r  =====================
__global__ void k_gates(const float* __restrict__ x, const float* __restrict__ Wb,
                        const float* __restrict__ Wg, const float* __restrict__ bg,
                        const float* __restrict__ Wh, const float* __restrict__ bh,
                        float* __restrict__ gates) {
  int t = blockIdx.x * blockDim.x + threadIdx.x;  // B*L*H threads
  if (t >= BHL) return;
  int b = t / (LLEN * NHEAD);
  int rem = t % (LLEN * NHEAD);
  int l = rem >> 2, h = rem & 3;
  const float* xr = x + (size_t)(b * LLEN + l) * DMODEL;
  float sbl = 0.f, sbg = 0.f, sg = 0.f, sh = 0.f;
  for (int d = 0; d < DMODEL; ++d) {
    float xv = xr[d];
    sbl += xv * Wb[d * 8 + h];
    sbg += xv * Wb[d * 8 + 4 + h];
    sg  += xv * Wg[d * 4 + h];
    sh  += xv * Wh[d * 4 + h];
  }
  sg += bg[h];
  sh += bh[h];
  float bl = 1.f / (1.f + expf(-sbl));
  float bgl = 1.f / (1.f + expf(-sbg));
  float m = fmaxf(fmaxf(sg, sh), 0.f);
  float e0 = expf(sg - m), e1 = expf(sh - m), e2 = expf(-m);
  float inv = 1.f / (e0 + e1 + e2);
  int gi = (b * NHEAD + h) * LLEN + l;
  gates[gi]           = bl;        // beta local
  gates[BHL + gi]     = bgl;       // beta global
  gates[2 * BHL + gi] = e0 * inv;  // w_l
  gates[3 * BHL + gi] = e1 * inv;  // w_g
  gates[4 * BHL + gi] = e2 * inv;  // w_r
}

// ======== conv(K=4, causal) + SiLU + per-head l2norm(q,k); v stays f32 =======
__global__ __launch_bounds__(256) void k_conv(
    const float* __restrict__ qp, const float* __restrict__ kp,
    const float* __restrict__ vp, const float* __restrict__ cq,
    const float* __restrict__ ck, const float* __restrict__ cv,
    bf16* __restrict__ qh, bf16* __restrict__ kh, float* __restrict__ vh) {
  int idx = blockIdx.x;
  int l = idx % LLEN;
  int h = (idx / LLEN) % NHEAD;
  int b = idx / (LLEN * NHEAD);
  int c = threadIdx.x;           // dk channel within head
  int ch = h * DKH + c;          // channel in D
  float tq = 0.f, tk = 0.f, tv = 0.f;
#pragma unroll
  for (int j = 0; j < CONVK; ++j) {
    int ls = l + j - (CONVK - 1);
    if (ls >= 0) {
      size_t off = (size_t)(b * LLEN + ls) * DMODEL + ch;
      tq += qp[off] * cq[ch * CONVK + j];
      tk += kp[off] * ck[ch * CONVK + j];
      tv += vp[off] * cv[ch * CONVK + j];
    }
  }
  tq = tq / (1.f + expf(-tq));
  tk = tk / (1.f + expf(-tk));
  tv = tv / (1.f + expf(-tv));
  __shared__ float r1[256], r2[256];
  r1[c] = tq * tq;
  r2[c] = tk * tk;
  __syncthreads();
  for (int s = 128; s > 0; s >>= 1) {
    if (c < s) { r1[c] += r1[c + s]; r2[c] += r2[c + s]; }
    __syncthreads();
  }
  float qn = tq * rsqrtf(r1[0] + 1e-12f);
  float kn = tk * rsqrtf(r2[0] + 1e-12f);
  size_t o = ((size_t)(b * NHEAD + h) * LLEN + l) * DKH + c;
  qh[o] = f2bf(qn);
  kh[o] = f2bf(kn);
  vh[o] = tv;
}

// ================= chunked delta-rule scan ==================================
// One block per (bh, branch, dv-slice). State S[dk=256, DVT=16] in LDS (f32).
__global__ __launch_bounds__(256) void k_scan(const bf16* __restrict__ qhall,
                                              const bf16* __restrict__ khall,
                                              const float* __restrict__ vhall,
                                              const float* __restrict__ gates,
                                              float* __restrict__ obr) {
  int bid = blockIdx.x;              // [0, 8*2*NDV)
  int dvi = bid % NDV;
  int br  = (bid / NDV) & 1;
  int bh  = bid / (NDV * 2);
  const bf16*  qg = qhall + (size_t)bh * LLEN * DKH;
  const bf16*  kg = khall + (size_t)bh * LLEN * DKH;
  const float* vg = vhall + (size_t)bh * LLEN * DVH + dvi * DVT;
  const float* beta = gates + (size_t)br * BHL + bh * LLEN;        // bl / bg
  const float* wrow = gates + (size_t)(2 + br) * BHL + bh * LLEN;  // wl / wg
  float* og = obr + ((size_t)(br * BBATCH * NHEAD + bh)) * LLEN * DVH + dvi * DVT;

  __shared__ float S[DKH * DVT];     // 16 KB  state (f32 master)
  __shared__ bf16  kbL[CCH * DKH];   // 16 KB  k*beta
  __shared__ bf16  wmn[CCH * DKH];   // 16 KB  -(T@kb)
  __shared__ float Tf[CCH * CCH];    //  4 KB  T (in-place inverse)
  __shared__ float uF[CCH * DVT];    //  2 KB  u
  __shared__ float vb[CCH * DVT];    //  2 KB  v*beta
  __shared__ float aA[CCH * CCH];    //  4 KB  causal(q@k^T)

  int tid = threadIdx.x, lane = tid & 31, w = tid >> 5;

  for (int i = tid; i < DKH * DVT; i += 256) S[i] = 0.f;
  __syncthreads();

  for (int chn = 0; chn < NCH; ++chn) {
    int l0 = chn * CCH;
    // ---- phase 0: stage kb = k*beta (bf16), vb = v*beta (f32) ----
    for (int i = tid; i < CCH * DKH; i += 256) {
      int r = i >> 8, cx = i & 255;
      kbL[i] = f2bf(bf2f(kg[(size_t)(l0 + r) * DKH + cx]) * beta[l0 + r]);
    }
    for (int i = tid; i < CCH * DVT; i += 256) {
      int r = i / DVT, j = i % DVT;
      vb[i] = vg[(size_t)(l0 + r) * DVH + j] * beta[l0 + r];
    }
    __syncthreads();

    // ---- phase 1: waves 0-3: T0 = strictlower(-(kb@k^T)); waves 4-7: a ----
    {
      int mt = (w >> 1) & 1, nt = w & 1;
      v8f acc = {};
      if (w < 4) {
#pragma unroll
        for (int kk = 0; kk < 8; ++kk) {
          v16bf fa = ld_a(kbL + mt * 16 * DKH + kk * 32, DKH, lane);
          v16bf fb = ld_bt(kg + (size_t)(l0 + nt * 16) * DKH + kk * 32, DKH, lane);
          acc = wmma_acc(fa, fb, acc);
        }
        int n = lane & 15, hk = (lane >> 4) & 1;
#pragma unroll
        for (int r = 0; r < 8; ++r) {
          int m = mt * 16 + r + hk * 8, nn = nt * 16 + n;
          Tf[m * CCH + nn] = (m > nn) ? -acc[r] : 0.f;
        }
      } else {
#pragma unroll
        for (int kk = 0; kk < 8; ++kk) {
          v16bf fa = ld_a(qg + (size_t)(l0 + mt * 16) * DKH + kk * 32, DKH, lane);
          v16bf fb = ld_bt(kg + (size_t)(l0 + nt * 16) * DKH + kk * 32, DKH, lane);
          acc = wmma_acc(fa, fb, acc);
        }
        int n = lane & 15, hk = (lane >> 4) & 1;
#pragma unroll
        for (int r = 0; r < 8; ++r) {
          int m = mt * 16 + r + hk * 8, nn = nt * 16 + n;
          aA[m * CCH + nn] = (nn <= m) ? acc[r] : 0.f;
        }
      }
    }
    __syncthreads();

    // ---- phase 2: forward-substitution inverse (wave 0, lane = column) ----
    if (w == 0) {
      volatile float* T = Tf;
      for (int i = 1; i < CCH; ++i) {
        float s = T[i * CCH + lane];
        for (int k2 = 0; k2 < i; ++k2) s += T[i * CCH + k2] * T[k2 * CCH + lane];
        if (lane < i) T[i * CCH + lane] = s;
      }
      T[lane * CCH + lane] = 1.0f;   // T += I
    }
    __syncthreads();

    // ---- phase 3: wmn = -(T@kb) (bf16), u = T@vb ----
    for (int job = w; job < 34; job += 8) {
      if (job < 32) {
        int mt = job & 1, nt = job >> 1;
        v8f acc = {};
        v16bf fa = ld_a_f32(Tf + mt * 16 * CCH, CCH, lane);
        v16bf fb = ld_b(kbL + nt * 16, DKH, lane);
        acc = wmma_acc(fa, fb, acc);
        int n = lane & 15, hk = (lane >> 4) & 1;
#pragma unroll
        for (int r = 0; r < 8; ++r) {
          int m = mt * 16 + r + hk * 8;
          wmn[m * DKH + nt * 16 + n] = f2bf(-acc[r]);
        }
      } else {
        int mt = job - 32;
        v8f acc = {};
        v16bf fa = ld_a_f32(Tf + mt * 16 * CCH, CCH, lane);
        v16bf fb = ld_b_f32(vb, DVT, lane);
        acc = wmma_acc(fa, fb, acc);
        int n = lane & 15, hk = (lane >> 4) & 1;
#pragma unroll
        for (int r = 0; r < 8; ++r) uF[(mt * 16 + r + hk * 8) * DVT + n] = acc[r];
      }
    }
    __syncthreads();

    // ---- phase 4: u = (u - wm@S) * w  -> uF ----
    if (w < 2) {
      int mt = w;
      v8f acc = ld_c(uF + mt * 16 * DVT, DVT, lane);
#pragma unroll
      for (int kk = 0; kk < 8; ++kk) {
        v16bf fa = ld_a(wmn + mt * 16 * DKH + kk * 32, DKH, lane);
        v16bf fb = ld_b_f32(S + kk * 32 * DVT, DVT, lane);
        acc = wmma_acc(fa, fb, acc);
      }
      int n = lane & 15, hk = (lane >> 4) & 1;
#pragma unroll
      for (int r = 0; r < 8; ++r) {
        int m = mt * 16 + r + hk * 8;
        uF[m * DVT + n] = acc[r] * wrow[l0 + m];
      }
    }
    __syncthreads();

    // ---- phase 5a: o = q@S + a@u  -> global ----
    if (w < 2) {
      int mt = w;
      v8f acc = {};
#pragma unroll
      for (int kk = 0; kk < 8; ++kk) {
        v16bf fa = ld_a(qg + (size_t)(l0 + mt * 16) * DKH + kk * 32, DKH, lane);
        v16bf fb = ld_b_f32(S + kk * 32 * DVT, DVT, lane);
        acc = wmma_acc(fa, fb, acc);
      }
      {
        v16bf fa = ld_a_f32(aA + mt * 16 * CCH, CCH, lane);
        v16bf fb = ld_b_f32(uF, DVT, lane);
        acc = wmma_acc(fa, fb, acc);
      }
      int n = lane & 15, hk = (lane >> 4) & 1;
#pragma unroll
      for (int r = 0; r < 8; ++r) {
        int m = mt * 16 + r + hk * 8;
        og[(size_t)(l0 + m) * DVH + n] = acc[r];
      }
    }
    __syncthreads();

    // ---- phase 5b: S += k^T @ u   (WMMA with C read-modify-write) ----
    for (int mt = w; mt < 16; mt += 8) {
      v8f acc = ld_c(S + mt * 16 * DVT, DVT, lane);
      v16bf fa = ld_at(kg + (size_t)l0 * DKH + mt * 16, DKH, lane);
      v16bf fb = ld_b_f32(uF, DVT, lane);
      acc = wmma_acc(fa, fb, acc);
      int n = lane & 15, hk = (lane >> 4) & 1;
#pragma unroll
      for (int r = 0; r < 8; ++r) {
        int m = mt * 16 + r + hk * 8;
        S[m * DVT + n] = acc[r];
      }
    }
    __syncthreads();
  }
}

// ================= epilogue: o = o_l + o_g + w_r*v, RMSNorm, -> bf16 ========
__global__ __launch_bounds__(256) void k_epi(const float* __restrict__ obr,
                                             const float* __restrict__ vh,
                                             const float* __restrict__ gates,
                                             const float* __restrict__ normw,
                                             bf16* __restrict__ ofin) {
  int idx = blockIdx.x;
  int l = idx % LLEN;
  int h = (idx / LLEN) % NHEAD;
  int b = idx / (LLEN * NHEAD);
  int c = threadIdx.x;
  int bh = b * NHEAD + h;
  size_t base = ((size_t)bh * LLEN + l) * DVH + c;
  size_t brstride = (size_t)BBATCH * NHEAD * LLEN * DVH;
  float wr = gates[4 * BHL + bh * LLEN + l];
  float o = obr[base] + obr[brstride + base] + wr * vh[base];
  __shared__ float r1[256];
  r1[c] = o * o;
  __syncthreads();
  for (int s = 128; s > 0; s >>= 1) {
    if (c < s) r1[c] += r1[c + s];
    __syncthreads();
  }
  float ms = r1[0] * (1.f / DVH) + 1e-5f;
  float out = o * rsqrtf(ms) * normw[c];
  ofin[((size_t)(b * LLEN + l)) * DMODEL + h * DVH + c] = f2bf(out);
}

// ============================ launcher ======================================
extern "C" void kernel_launch(void* const* d_in, const int* in_sizes, int n_in,
                              void* d_out, int out_size, void* d_ws, size_t ws_size,
                              hipStream_t stream) {
  const float* x     = (const float*)d_in[0];
  const float* Wq    = (const float*)d_in[1];
  const float* Wk    = (const float*)d_in[2];
  const float* Wv    = (const float*)d_in[3];
  const float* convq = (const float*)d_in[4];
  const float* convk = (const float*)d_in[5];
  const float* convv = (const float*)d_in[6];
  const float* Wb    = (const float*)d_in[7];
  const float* Wg    = (const float*)d_in[8];
  const float* bg    = (const float*)d_in[9];
  const float* Wh    = (const float*)d_in[10];
  const float* bh    = (const float*)d_in[11];
  const float* normw = (const float*)d_in[12];
  const float* Wo    = (const float*)d_in[13];

  char* ws = (char*)d_ws;
  const size_t MB = 1ull << 20;
  float* q_pre = (float*)(ws + 0 * MB);    // 16 MB  [B*L, D]
  float* k_pre = (float*)(ws + 16 * MB);   // 16 MB
  float* v_pre = (float*)(ws + 32 * MB);   // 16 MB
  bf16*  qh    = (bf16*)(ws + 48 * MB);    //  8 MB  [B,H,L,dk]
  bf16*  kh    = (bf16*)(ws + 56 * MB);    //  8 MB
  float* vh    = (float*)(ws + 64 * MB);   // 16 MB  [B,H,L,dv]
  float* gates = (float*)(ws + 80 * MB);   // 320 KB (5 x B*H*L)
  float* obr   = (float*)(ws + 81 * MB);   // 32 MB  [2,B,H,L,dv]
  bf16*  ofin  = (bf16*)(ws + 113 * MB);   //  8 MB  [B*L, D]

  dim3 blk(256);
  dim3 gproj(BL / 128, DMODEL / 64);  // (32, 16)

  k_gemm<false><<<gproj, blk, 0, stream>>>((const void*)x, Wq, q_pre, BL, DMODEL, DMODEL);
  k_gemm<false><<<gproj, blk, 0, stream>>>((const void*)x, Wk, k_pre, BL, DMODEL, DMODEL);
  k_gemm<false><<<gproj, blk, 0, stream>>>((const void*)x, Wv, v_pre, BL, DMODEL, DMODEL);

  k_gates<<<dim3((BHL + 255) / 256), blk, 0, stream>>>(x, Wb, Wg, bg, Wh, bh, gates);

  k_conv<<<dim3(BHL), blk, 0, stream>>>(q_pre, k_pre, v_pre, convq, convk, convv,
                                        qh, kh, vh);

  k_scan<<<dim3(BBATCH * NHEAD * 2 * NDV), blk, 0, stream>>>(qh, kh, vh, gates, obr);

  k_epi<<<dim3(BHL), blk, 0, stream>>>(obr, vh, gates, normw, ofin);

  k_gemm<true><<<gproj, blk, 0, stream>>>((const void*)ofin, Wo, (float*)d_out,
                                          BL, DMODEL, DMODEL);
}
